// TransformerDecoder2D_42082089566460
// MI455X (gfx1250) — matrix operand order repaired and checked by reading, not compiled, not run on hardware
//
#include <hip/hip_runtime.h>
#include <hip/hip_bf16.h>

// ---------------------------------------------------------------------------
// Sizes (fixed by reference setup_inputs)
// ---------------------------------------------------------------------------
#define LL   8
#define DD   1024
#define HH   16
#define DFFc 4096
#define AAc  4096
#define SS   66
#define TT   67            // S + 1 (pool token)
#define BBc  256
#define DKc  64
#define BIAS_LD 78         // rel_bias trailing dims are 78x78, sliced to T

typedef __attribute__((ext_vector_type(16))) _Float16 v16h;
typedef __attribute__((ext_vector_type(8)))  _Float16 v8h;
typedef __attribute__((ext_vector_type(8)))  float    v8f;
typedef __attribute__((ext_vector_type(4)))  unsigned int u32x4;
typedef __attribute__((ext_vector_type(8)))  int          i32x8;
typedef __attribute__((ext_vector_type(4)))  int          i32x4;

// ---------------------------------------------------------------------------
// f32 -> f16 conversion (weights / activations)
// ---------------------------------------------------------------------------
__global__ __launch_bounds__(256)
void cvt_f32_f16(const float* __restrict__ in, _Float16* __restrict__ out, size_t n) {
    size_t i = (size_t)blockIdx.x * 256 + threadIdx.x;
    size_t stride = (size_t)gridDim.x * 256;
    for (; i < n; i += stride) out[i] = (_Float16)in[i];
}

// ---------------------------------------------------------------------------
// Embedding * sqrt(D) + pool token + sinusoidal positional encoding
// ---------------------------------------------------------------------------
__global__ __launch_bounds__(256)
void embed_kernel(const int* __restrict__ x, const float* __restrict__ emb,
                  const float* __restrict__ pool,
                  float* __restrict__ h, _Float16* __restrict__ h16) {
    const int bt = blockIdx.x;
    const int b  = bt / TT;
    const int t  = bt % TT;
    const int tid = threadIdx.x;
    const float sqrtD = 32.0f;                     // sqrt(1024)
    const float negln = -logf(10000.0f);
    int tok = (t == 0) ? 0 : x[b * SS + (t - 1)];
    #pragma unroll
    for (int i = 0; i < 4; ++i) {
        int d = tid + 256 * i;
        float base = (t == 0) ? pool[d] : emb[(size_t)tok * DD + d] * sqrtD;
        int pair = d >> 1;
        float freq = __expf(negln * (float)(2 * pair) / (float)DD);
        float ang  = (float)t * freq;
        float pe   = (d & 1) ? cosf(ang) : sinf(ang);
        float val  = base + pe;
        h  [(size_t)bt * DD + d] = val;
        h16[(size_t)bt * DD + d] = (_Float16)val;
    }
}

// ---------------------------------------------------------------------------
// WMMA GEMM: C[M,N] = A[M,K](f16) * B[K,N](f16) (+bias) (+exact GELU)
// 128x128 block tile, 8 waves, each wave 32(M) x 64(N), K-step 32.
// A tile is DMA'd by the Tensor Data Mover (tensor_load_to_lds) with the
// hardware pad engine producing the 40-half padded LDS rows; OOB rows are
// zero-filled by the TDM. B tile is staged transposed by the 8 waves in
// parallel with the DMA. Emits v_wmma_f32_16x16x32_f16 + tensor_load_to_lds
// + s_wait_tensorcnt + global_prefetch_b8.
// ---------------------------------------------------------------------------
template<bool BIAS, bool GELU, bool WF32, bool WF16>
__global__ __launch_bounds__(256)
void gemm_wmma_f16(const _Float16* __restrict__ A, const _Float16* __restrict__ Bw,
                   const float* __restrict__ bias,
                   float* __restrict__ Cf, _Float16* __restrict__ Ch,
                   int M, int N, int K) {
    __shared__ _Float16 sA[128][40];   // 128 rows (M) x 32 halves (K), pad 8
    __shared__ _Float16 sB[128][40];   // 128 rows (N) x 32 halves (K), pad 8

    const int tid    = threadIdx.x;
    const int wave   = tid >> 5;
    const int lane   = tid & 31;
    const int lh     = lane & 15;         // 0..15
    const int hiHalf = (lane >> 4);       // 0 or 1
    const int akoff  = hiHalf << 3;       // A: K offset 0 or 8
    const int bkoff  = hiHalf << 3;       // B: K offset 0 or 8

    const int m0 = blockIdx.y * 128;
    const int n0 = blockIdx.x * 128;

    const int waveM = wave >> 1;          // 0..3 -> 32 rows each
    const int waveN = wave & 1;           // 0..1 -> 64 cols each

    // --- invariant parts of the Tensor DMA descriptor (uniform) -----------
    const unsigned int ldsA = (unsigned int)(uintptr_t)&sA[0][0];
    // group1 word0: data_size=2B (code 1), pad_enable, pad_interval=16dw
    // (code 3), pad_amount=4dw (code 3)  ->  64B tile row + 16B pad = 80B
    const unsigned int g1w0 = (1u << 16) | (1u << 20) | (3u << 22) | (3u << 25);

    v8f acc[2][4] = {};

    for (int k0 = 0; k0 < K; k0 += 32) {
        // ---- wave 0: issue TDM load of the 128x32 A tile ------------------
        if (wave == 0) {
            unsigned long long ga =
                (unsigned long long)(uintptr_t)A +
                ((unsigned long long)m0 * (unsigned long long)K + (unsigned long long)k0) * 2ull;
            u32x4 g0;
            g0[0] = 1u;                                   // count=1 (valid D#)
            g0[1] = ldsA;                                 // lds_addr (bytes)
            g0[2] = (unsigned int)(ga & 0xFFFFFFFFu);     // global_addr[31:0]
            g0[3] = (unsigned int)((ga >> 32) & 0x01FFFFFFu) | (2u << 30); // type=2
            i32x8 g1;
            g1[0] = (int)g1w0;
            g1[1] = (int)(((unsigned)K & 0xFFFFu) << 16);                 // dim0[15:0]
            g1[2] = (int)((((unsigned)K >> 16) & 0xFFFFu) |
                          (((unsigned)M & 0xFFFFu) << 16));               // dim0 hi | dim1 lo
            g1[3] = (int)((((unsigned)M >> 16) & 0xFFFFu) | (32u << 16)); // dim1 hi | tile_dim0=32
            g1[4] = (int)128u;                                            // tile_dim1=128, tile_dim2=0
            g1[5] = (int)(unsigned)K;                                     // dim0_stride[31:0]
            g1[6] = 0;                                                    // stride hi
            g1[7] = 0;
            i32x4 z4 = {};
#if defined(__clang_major__) && (__clang_major__ >= 23)
            i32x8 z8 = {};
            __builtin_amdgcn_tensor_load_to_lds(g0, g1, z4, z4, z8, 0);
#else
            __builtin_amdgcn_tensor_load_to_lds(g0, g1, z4, z4, 0);
#endif
        }

        // ---- all 256 threads: stage B tile transposed sB[n][k] ------------
        // both coalesced v8h global loads issued before the LDS scatter so
        // the two VMEM ops are in flight together
        const int kk0   = (tid * 2) >> 4;         // 0..31
        const int nseg0 = (tid * 2) & 15;         // even segment
        v8h bv0 = *(const v8h*)(Bw + (size_t)(k0 + kk0) * N + n0 + nseg0 * 8);
        v8h bv1 = *(const v8h*)(Bw + (size_t)(k0 + kk0) * N + n0 + (nseg0 + 1) * 8);
        #pragma unroll
        for (int e = 0; e < 8; ++e) sB[nseg0 * 8 + e][kk0]       = bv0[e];
        #pragma unroll
        for (int e = 0; e < 8; ++e) sB[(nseg0 + 1) * 8 + e][kk0] = bv1[e];

        // gfx1250 prefetch of next B tile (global_prefetch_b8, WGP scope)
        if (k0 + 32 < K)
            __builtin_prefetch(Bw + (size_t)(k0 + 32 + (tid >> 4)) * N + n0 + (tid & 15) * 8, 0, 3);

        if (wave == 0) __builtin_amdgcn_s_wait_tensorcnt(0);
        __syncthreads();

        // ---- A fragments (ISA 16-bit A 16x32 layout) ----------------------
        v16h afrag[2];
        #pragma unroll
        for (int mt = 0; mt < 2; ++mt) {
            int row = waveM * 32 + mt * 16 + lh;
            v8h lo = *(const v8h*)(&sA[row][akoff]);        // K = akoff..akoff+7
            v8h hi = *(const v8h*)(&sA[row][akoff + 16]);   // K = akoff+16..+23
            #pragma unroll
            for (int e = 0; e < 8; ++e) { afrag[mt][e] = lo[e]; afrag[mt][e + 8] = hi[e]; }
        }
        // ---- B fragments (lane<16: K 0..7,16..23, lane>=16: K 8..15,24..31) + 8 WMMAs -
        #pragma unroll
        for (int nt = 0; nt < 4; ++nt) {
            int col = waveN * 64 + nt * 16 + lh;
            v8h lo = *(const v8h*)(&sB[col][bkoff]);
            v8h hi = *(const v8h*)(&sB[col][bkoff + 16]);
            v16h bfrag;
            #pragma unroll
            for (int e = 0; e < 8; ++e) { bfrag[e] = lo[e]; bfrag[e + 8] = hi[e]; }
            #pragma unroll
            for (int mt = 0; mt < 2; ++mt) {
                acc[mt][nt] = __builtin_amdgcn_wmma_f32_16x16x32_f16(
                    false, afrag[mt], false, bfrag, (short)0, acc[mt][nt], false, false);
            }
        }
        __syncthreads();
    }

    // ---- epilogue: C/D f32 layout = VGPR r -> M=r (lanes 0-15), M=r+8 (16-31)
    const int rowHi = hiHalf << 3;
    #pragma unroll
    for (int mt = 0; mt < 2; ++mt) {
        #pragma unroll
        for (int nt = 0; nt < 4; ++nt) {
            int gn = n0 + waveN * 64 + nt * 16 + lh;
            #pragma unroll
            for (int r = 0; r < 8; ++r) {
                int gm = m0 + waveM * 32 + mt * 16 + r + rowHi;
                if (gm < M) {
                    float val = acc[mt][nt][r];
                    if (BIAS) val += bias[gn];
                    if (GELU) val = 0.5f * val * (1.0f + erff(val * 0.70710678118654752f));
                    if (WF32) Cf[(size_t)gm * N + gn] = val;
                    if (WF16) Ch[(size_t)gm * N + gn] = (_Float16)val;
                }
            }
        }
    }
}

// ---------------------------------------------------------------------------
// Attention: one 64-thread block per (query t, head, batch)
// scores + rel_bias slice -> softmax -> weighted V. (~2% of FLOPs; VALU.)
// ---------------------------------------------------------------------------
__global__ __launch_bounds__(64)
void attn_kernel(const float* __restrict__ q, const float* __restrict__ k,
                 const float* __restrict__ v, const float* __restrict__ bias,
                 float* __restrict__ ctx) {
    const int tq = blockIdx.x, hh = blockIdx.y, b = blockIdx.z;
    const int tid = threadIdx.x;
    __shared__ float sc[80];
    __shared__ float sinv;
    const float scale = 0.125f;  // 1/sqrt(64)

    const float* qp = q + ((size_t)(b * TT + tq)) * DD + hh * DKc;
    for (int kk = tid; kk < TT; kk += 64) {
        const float* kp = k + ((size_t)(b * TT + kk)) * DD + hh * DKc;
        float dot = 0.f;
        #pragma unroll
        for (int d = 0; d < DKc; ++d) dot += qp[d] * kp[d];
        sc[kk] = dot * scale + bias[(size_t)hh * BIAS_LD * BIAS_LD + tq * BIAS_LD + kk];
    }
    __syncthreads();
    if (tid == 0) {
        float m = -1e30f;
        for (int kk = 0; kk < TT; ++kk) m = fmaxf(m, sc[kk]);
        float s = 0.f;
        for (int kk = 0; kk < TT; ++kk) { float e = __expf(sc[kk] - m); sc[kk] = e; s += e; }
        sinv = 1.f / s;
    }
    __syncthreads();
    float inv = sinv;
    float out = 0.f;
    for (int kk = 0; kk < TT; ++kk)
        out += sc[kk] * v[((size_t)(b * TT + kk)) * DD + hh * DKc + tid];
    ctx[((size_t)(b * TT + tq)) * DD + hh * DKc + tid] = out * inv;
}

// ---------------------------------------------------------------------------
// Fused residual add + LayerNorm; writes f32 + f16. One block per row.
// ---------------------------------------------------------------------------
__global__ __launch_bounds__(256)
void add_ln_kernel(const float* __restrict__ resid, const float* __restrict__ delta,
                   const float* __restrict__ g, const float* __restrict__ bb,
                   float* __restrict__ hout, _Float16* __restrict__ h16out) {
    const int row = blockIdx.x;
    const int tid = threadIdx.x;
    __shared__ float ssum[256];
    __shared__ float ssq[256];
    float x[4];
    const float* rp = resid + (size_t)row * DD;
    const float* dp = delta + (size_t)row * DD;
    float s = 0.f, s2 = 0.f;
    #pragma unroll
    for (int i = 0; i < 4; ++i) {
        int d = tid + 256 * i;
        x[i] = rp[d] + dp[d];
        s += x[i]; s2 += x[i] * x[i];
    }
    ssum[tid] = s; ssq[tid] = s2;
    __syncthreads();
    for (int off = 128; off > 0; off >>= 1) {
        if (tid < off) { ssum[tid] += ssum[tid + off]; ssq[tid] += ssq[tid + off]; }
        __syncthreads();
    }
    float mean = ssum[0] * (1.0f / DD);
    float var  = ssq[0] * (1.0f / DD) - mean * mean;
    float rstd = rsqrtf(var + 1e-5f);
    #pragma unroll
    for (int i = 0; i < 4; ++i) {
        int d = tid + 256 * i;
        float y = (x[i] - mean) * rstd * g[d] + bb[d];
        hout  [(size_t)row * DD + d] = y;
        h16out[(size_t)row * DD + d] = (_Float16)y;
    }
}

// ---------------------------------------------------------------------------
// Gather pool-token rows (t==0) into dense [B, D] f16 matrix
// ---------------------------------------------------------------------------
__global__ __launch_bounds__(256)
void gather_pool_kernel(const _Float16* __restrict__ h16, _Float16* __restrict__ out) {
    int b = blockIdx.x;
    int d = threadIdx.x + blockIdx.y * 256;
    out[(size_t)b * DD + d] = h16[(size_t)b * TT * DD + d];
}

// ---------------------------------------------------------------------------
// Host orchestration
// ---------------------------------------------------------------------------
extern "C" void kernel_launch(void* const* d_in, const int* in_sizes, int n_in,
                              void* d_out, int out_size, void* d_ws, size_t ws_size,
                              hipStream_t stream) {
    const int*   x    = (const int*)  d_in[0];
    const float* emb  = (const float*)d_in[1];
    const float* pool = (const float*)d_in[2];
    const float* Wq   = (const float*)d_in[3];
    const float* Wk   = (const float*)d_in[4];
    const float* Wv   = (const float*)d_in[5];
    const float* Wo   = (const float*)d_in[6];
    const float* relb = (const float*)d_in[7];
    const float* ln1g = (const float*)d_in[8];
    const float* ln1b = (const float*)d_in[9];
    const float* ln2g = (const float*)d_in[10];
    const float* ln2b = (const float*)d_in[11];
    const float* W1   = (const float*)d_in[12];
    const float* b1   = (const float*)d_in[13];
    const float* W2   = (const float*)d_in[14];
    const float* b2   = (const float*)d_in[15];
    const float* Wout = (const float*)d_in[16];
    const float* bout = (const float*)d_in[17];
    float* out        = (float*)d_out;

    const size_t BT = (size_t)BBc * TT;     // 17152 (multiple of 128)

    char* wp = (char*)d_ws;
    auto alloc = [&](size_t bytes) -> void* {
        void* p = (void*)wp;
        wp += (bytes + 255) & ~(size_t)255;
        return p;
    };

    float*    h      = (float*)   alloc(BT * DD * 4);
    _Float16* h16    = (_Float16*)alloc(BT * DD * 2);
    float*    qb     = (float*)   alloc(BT * DD * 4);
    float*    kb     = (float*)   alloc(BT * DD * 4);
    float*    vb     = (float*)   alloc(BT * DD * 4);
    float*    ctx    = (float*)   alloc(BT * DD * 4);
    _Float16* ctx16  = (_Float16*)alloc(BT * DD * 2);
    float*    tmp    = (float*)   alloc(BT * DD * 4);
    _Float16* ffn16  = (_Float16*)alloc(BT * DFFc * 2);
    _Float16* pool16 = (_Float16*)alloc((size_t)BBc * DD * 2);
    _Float16* wq16   = (_Float16*)alloc((size_t)LL * DD * DD * 2);
    _Float16* wk16   = (_Float16*)alloc((size_t)LL * DD * DD * 2);
    _Float16* wv16   = (_Float16*)alloc((size_t)LL * DD * DD * 2);
    _Float16* wo16   = (_Float16*)alloc((size_t)LL * DD * DD * 2);
    _Float16* w116   = (_Float16*)alloc((size_t)LL * DD * DFFc * 2);
    _Float16* w216   = (_Float16*)alloc((size_t)LL * DFFc * DD * 2);
    _Float16* wout16 = (_Float16*)alloc((size_t)DD * AAc * 2);

    // ---- one-time (per launch) f16 weight conversion
    const size_t nDD = (size_t)LL * DD * DD;
    cvt_f32_f16<<<4096, 256, 0, stream>>>(Wq,   wq16,   nDD);
    cvt_f32_f16<<<4096, 256, 0, stream>>>(Wk,   wk16,   nDD);
    cvt_f32_f16<<<4096, 256, 0, stream>>>(Wv,   wv16,   nDD);
    cvt_f32_f16<<<4096, 256, 0, stream>>>(Wo,   wo16,   nDD);
    cvt_f32_f16<<<4096, 256, 0, stream>>>(W1,   w116,   (size_t)LL * DD * DFFc);
    cvt_f32_f16<<<4096, 256, 0, stream>>>(W2,   w216,   (size_t)LL * DFFc * DD);
    cvt_f32_f16<<<4096, 256, 0, stream>>>(Wout, wout16, (size_t)DD * AAc);

    // ---- embedding + positional encoding
    embed_kernel<<<(int)BT, 256, 0, stream>>>(x, emb, pool, h, h16);

    const dim3 gD  (DD   / 128, (int)(BT / 128));   // N=1024 GEMMs over all tokens
    const dim3 gDFF(DFFc / 128, (int)(BT / 128));   // N=4096 GEMM (FFN up)
    const dim3 gAttn(TT, HH, BBc);
    const int  ln_rows = (int)BT;

    for (int i = 0; i < LL; ++i) {
        const size_t oDD = (size_t)i * DD * DD;
        // Q, K, V projections (WMMA + TDM)
        gemm_wmma_f16<false, false, true,  false><<<gD, 256, 0, stream>>>(
            h16, wq16 + oDD, nullptr, qb, nullptr, (int)BT, DD, DD);
        gemm_wmma_f16<false, false, true,  false><<<gD, 256, 0, stream>>>(
            h16, wk16 + oDD, nullptr, kb, nullptr, (int)BT, DD, DD);
        gemm_wmma_f16<false, false, true,  false><<<gD, 256, 0, stream>>>(
            h16, wv16 + oDD, nullptr, vb, nullptr, (int)BT, DD, DD);
        // attention core
        attn_kernel<<<gAttn, 64, 0, stream>>>(
            qb, kb, vb, relb + (size_t)i * HH * BIAS_LD * BIAS_LD, ctx);
        // O projection (WMMA)
        cvt_f32_f16<<<4096, 256, 0, stream>>>(ctx, ctx16, BT * DD);
        gemm_wmma_f16<false, false, true,  false><<<gD, 256, 0, stream>>>(
            ctx16, wo16 + oDD, nullptr, tmp, nullptr, (int)BT, DD, DD);
        // residual + LN1
        add_ln_kernel<<<ln_rows, 256, 0, stream>>>(
            h, tmp, ln1g + (size_t)i * DD, ln1b + (size_t)i * DD, h, h16);
        // FFN up + GELU (WMMA, f16 out only)
        gemm_wmma_f16<true,  true,  false, true ><<<gDFF, 256, 0, stream>>>(
            h16, w116 + (size_t)i * DD * DFFc, b1 + (size_t)i * DFFc,
            nullptr, ffn16, (int)BT, DFFc, DD);
        // FFN down (WMMA)
        gemm_wmma_f16<true,  false, true,  false><<<gD, 256, 0, stream>>>(
            ffn16, w216 + (size_t)i * DFFc * DD, b2 + (size_t)i * DD,
            tmp, nullptr, (int)BT, DD, DFFc);
        // residual + LN2
        add_ln_kernel<<<ln_rows, 256, 0, stream>>>(
            h, tmp, ln2g + (size_t)i * DD, ln2b + (size_t)i * DD, h, h16);
    }

    // ---- pooled token -> output projection (WMMA)
    gather_pool_kernel<<<dim3(BBc, DD / 256), 256, 0, stream>>>(h16, pool16);
    gemm_wmma_f16<true, false, true, false><<<dim3(AAc / 128, BBc / 128), 256, 0, stream>>>(
        pool16, wout16, bout, out, nullptr, BBc, AAc, DD);
}
